// Grapher_23888608100911
// MI455X (gfx1250) — matrix operand order repaired and checked by reading, not compiled, not run on hardware
//
#include <hip/hip_runtime.h>
#include <hip/hip_bf16.h>

// ---------------------------------------------------------------------------
// Fused FFC/Grapher block for MI455X (gfx1250, wave32, WMMA bf16).
// All bf16 intermediates are NHWC (channel-last) so WMMA B-fragments and
// GEMM epilogues use 128-bit vector memory ops.
//   stage0: depthwise 7x7 + bias + residual          -> y  (bf16 NHWC)
//   stage1: fc1 1x1 GEMM (K=192) + BN                -> x1 (bf16 NHWC) [WMMA]
//   stage2: MRConv4d(x1) fused into gc GEMM (K=384)
//           + BN + exact GELU                        -> x2 (bf16 NHWC) [WMMA]
//   stage3: fc2 1x1 GEMM (K=192) + BN                -> out (f32 NCHW) [WMMA]
// ---------------------------------------------------------------------------

typedef __attribute__((ext_vector_type(16))) __bf16 v16bf;
typedef __attribute__((ext_vector_type(8)))  float  v8f;

union BF16x16 {
    v16bf v;
    unsigned short u[16];
    unsigned int   w[8];
    uint4          q[2];
};

__device__ __forceinline__ unsigned short f2bf(float f) {
    unsigned int u = __float_as_uint(f);
    unsigned int r = u + 0x7FFFu + ((u >> 16) & 1u);   // round-to-nearest-even
    return (unsigned short)(r >> 16);
}
__device__ __forceinline__ float bf2f(unsigned short h) {
    return __uint_as_float(((unsigned int)h) << 16);
}

#define CCH  192
#define HWP  3136   // 56*56
#define WIMG 56

// ---------------------------------------------------------------------------
// Convert the three 1x1-conv weight matrices to bf16 (concatenated region).
// ---------------------------------------------------------------------------
__global__ __launch_bounds__(256)
void cvt_weights(const float* __restrict__ w1, const float* __restrict__ wg,
                 const float* __restrict__ w2, unsigned short* __restrict__ out) {
    int i = blockIdx.x * 256 + threadIdx.x;
    if (i >= 147456) return;
    float f;
    if (i < 36864)        f = w1[i];
    else if (i < 110592)  f = wg[i - 36864];
    else                  f = w2[i - 110592];
    out[i] = f2bf(f);
}

// ---------------------------------------------------------------------------
// Fuse conv-bias + BatchNorm into per-channel scale/shift (3 layers x 192).
// ---------------------------------------------------------------------------
__global__ __launch_bounds__(576)
void bn_consts(const float* cb0, const float* g0, const float* be0, const float* m0, const float* v0,
               const float* cb1, const float* g1, const float* be1, const float* m1, const float* v1,
               const float* cb2, const float* g2, const float* be2, const float* m2, const float* v2,
               float* __restrict__ s_all, float* __restrict__ t_all) {
    int i = threadIdx.x;
    if (i >= 576) return;
    int layer = i / CCH, c = i % CCH;
    const float *cb, *g, *be, *m, *v;
    if (layer == 0)      { cb = cb0; g = g0; be = be0; m = m0; v = v0; }
    else if (layer == 1) { cb = cb1; g = g1; be = be1; m = m1; v = v1; }
    else                 { cb = cb2; g = g2; be = be2; m = m2; v = v2; }
    float inv = g[c] * rsqrtf(v[c] + 1e-5f);
    s_all[i] = inv;
    t_all[i] = (cb[c] - m[c]) * inv + be[c];
}

// ---------------------------------------------------------------------------
// Depthwise 7x7 (SAME, zero pad) + bias + residual.
// Channel-fastest thread order -> NHWC bf16 stores fully coalesced.
// ---------------------------------------------------------------------------
__global__ __launch_bounds__(256)
void dwconv7(const float* __restrict__ x, const float* __restrict__ w,
             const float* __restrict__ bias, unsigned short* __restrict__ y, int total) {
    int idx = blockIdx.x * 256 + threadIdx.x;      // idx = (b*HW + hw)*C + c  (NHWC order)
    if (idx >= total) return;
    int c    = idx % CCH;
    int rest = idx / CCH;
    int hw   = rest % HWP;
    int b    = rest / HWP;
    int h = hw / WIMG, ww = hw % WIMG;
    const float* xp = x + ((size_t)b * CCH + c) * HWP;   // NCHW input plane
    const float* wc = w + c * 49;
    float acc = bias[c] + xp[hw];
#pragma unroll
    for (int kh = 0; kh < 7; ++kh) {
        int hy = h + kh - 3;
        if (hy < 0 || hy >= WIMG) continue;
#pragma unroll
        for (int kw = 0; kw < 7; ++kw) {
            int wx = ww + kw - 3;
            if (wx < 0 || wx >= WIMG) continue;
            acc = fmaf(xp[hy * WIMG + wx], wc[kh * 7 + kw], acc);
        }
    }
    y[idx] = f2bf(acc);
}

// ---------------------------------------------------------------------------
// 1x1-conv GEMM via v_wmma_f32_16x16x32_bf16, NHWC activations.
//  block = 128 threads (4 waves); tile = 64(M) x 64(N); grid=(49, 3, 32).
//  All 4 B-fragments are loaded into distinct registers before the WMMA
//  burst so the 8 b128 loads overlap (staggered s_wait_loadcnt instead of
//  load->wait0->wmma serialization).
//  MODE 0: plain B,                    BN,       bf16 NHWC out (fc1)
//  MODE 1: B = [x1 ; MRConv4d(x1)],    BN+GELU,  bf16 NHWC out (gc, K=384)
//  MODE 2: plain B,                    BN,       f32 NCHW out  (fc2)
// ---------------------------------------------------------------------------
template <int MODE, int KD>
__global__ __launch_bounds__(128)
void gemm1x1(const unsigned short* __restrict__ Wb,   // [192, KD] bf16 row-major
             const unsigned short* __restrict__ act,  // bf16 NHWC [B,HW,192]
             const float* __restrict__ s, const float* __restrict__ t,
             unsigned short* __restrict__ outBf,      // bf16 NHWC
             float* __restrict__ outF) {              // f32 NCHW
    const int lane = threadIdx.x & 31;
    const int wave = threadIdx.x >> 5;
    const int half = lane >> 4;
    const int l15  = lane & 15;
    const int nBase   = blockIdx.x * 64;
    const int rowBase = blockIdx.y * 64 + wave * 16;
    const int b       = blockIdx.z;
    const size_t pixBase = (size_t)b * HWP;

    v8f acc[4] = {};

    // per-lane pixel index and NHWC bases (+ wrapped neighbors for MRConv)
    int    n[4];
    size_t pC[4], pU[4], pD[4], pL[4], pR[4];
#pragma unroll
    for (int sB = 0; sB < 4; ++sB) {
        int nn = nBase + sB * 16 + l15;
        n[sB]  = nn;
        pC[sB] = (pixBase + nn) * CCH;
        if (MODE == 1) {
            int h = nn / WIMG, w0 = nn % WIMG;
            pU[sB] = (pixBase + ((h + WIMG - 2) % WIMG) * WIMG + w0) * CCH;  // roll(+K): x[h-2]
            pD[sB] = (pixBase + ((h + 2) % WIMG) * WIMG + w0) * CCH;         // roll(-K): x[h+2]
            pL[sB] = (pixBase + h * WIMG + ((w0 + WIMG - 2) % WIMG)) * CCH;
            pR[sB] = (pixBase + h * WIMG + ((w0 + 2) % WIMG)) * CCH;
        }
    }

    const unsigned short* wrow = Wb + (size_t)(rowBase + l15) * KD;
    constexpr int nsteps = KD >> 5;

#pragma unroll 2
    for (int ks = 0; ks < nsteps; ++ks) {
        // ---- A fragment: 16x32 bf16, contiguous K = 16*half + s; two 16B chunks per lane ----
        BF16x16 a;
        a.q[0] = *(const uint4*)(wrow + ks * 32 + half * 16);       // K: 0-7  / 16-23
        a.q[1] = *(const uint4*)(wrow + ks * 32 + half * 16 + 8);   // K: 8-15 / 24-31

        const int kOff  = ks * 32 + half * 16;            // this lane's 16-K chunk
        const bool isXj = (MODE == 1) && (kOff >= CCH);   // never straddles a half

        // ---- phase 1: all four B fragments into distinct registers ----
        BF16x16 bm[4];
        if (!isXj) {
#pragma unroll
            for (int sB = 0; sB < 4; ++sB) {
                const unsigned short* p = act + pC[sB] + kOff;
                bm[sB].q[0] = *(const uint4*)(p);
                bm[sB].q[1] = *(const uint4*)(p + 8);
            }
        } else {
#pragma unroll
            for (int sB = 0; sB < 4; ++sB) {
                const int c0 = kOff - CCH;
                BF16x16 ctr, nu, nd, nl, nr;
                ctr.q[0] = *(const uint4*)(act + pC[sB] + c0);
                ctr.q[1] = *(const uint4*)(act + pC[sB] + c0 + 8);
                nu.q[0]  = *(const uint4*)(act + pU[sB] + c0);
                nu.q[1]  = *(const uint4*)(act + pU[sB] + c0 + 8);
                nd.q[0]  = *(const uint4*)(act + pD[sB] + c0);
                nd.q[1]  = *(const uint4*)(act + pD[sB] + c0 + 8);
                nl.q[0]  = *(const uint4*)(act + pL[sB] + c0);
                nl.q[1]  = *(const uint4*)(act + pL[sB] + c0 + 8);
                nr.q[0]  = *(const uint4*)(act + pR[sB] + c0);
                nr.q[1]  = *(const uint4*)(act + pR[sB] + c0 + 8);
#pragma unroll
                for (int i = 0; i < 16; ++i) {
                    float cv = bf2f(ctr.u[i]);
                    float d  = bf2f(nu.u[i]) - cv;
                    d = fmaxf(d, bf2f(nd.u[i]) - cv);
                    d = fmaxf(d, bf2f(nl.u[i]) - cv);
                    d = fmaxf(d, bf2f(nr.u[i]) - cv);
                    d = fmaxf(d, 0.0f);                    // x_j
                    bm[sB].u[i] = f2bf(d);
                }
            }
        }

        // ---- phase 2: WMMA burst (loads above drain while these issue) ----
#pragma unroll
        for (int sB = 0; sB < 4; ++sB)
            acc[sB] = __builtin_amdgcn_wmma_f32_16x16x32_bf16(
                false, a.v, false, bm[sB].v, (short)0, acc[sB], false, false);
    }

    // ---- epilogue: BN (+GELU); packed 16B NHWC stores (or f32 NCHW) ----
    const int m0 = rowBase + 8 * half;                    // 8 consecutive channels
#pragma unroll
    for (int sB = 0; sB < 4; ++sB) {
        union { unsigned short u[8]; uint4 q; } o;
#pragma unroll
        for (int r = 0; r < 8; ++r) {
            int m   = m0 + r;
            float v = acc[sB][r] * s[m] + t[m];
            if (MODE == 1)
                v = 0.5f * v * (1.0f + erff(v * 0.70710678118654752f));  // exact GELU
            if (MODE == 2)
                outF[((size_t)b * CCH + m) * HWP + n[sB]] = v;
            else
                o.u[r] = f2bf(v);
        }
        if (MODE != 2)
            *(uint4*)(outBf + pC[sB] + m0) = o.q;
    }
}

// ---------------------------------------------------------------------------
extern "C" void kernel_launch(void* const* d_in, const int* in_sizes, int n_in,
                              void* d_out, int out_size, void* d_ws, size_t ws_size,
                              hipStream_t stream) {
    const float* x     = (const float*)d_in[0];
    const float* cpe_w = (const float*)d_in[1];
    const float* cpe_b = (const float*)d_in[2];
    const float* fc1_w = (const float*)d_in[3];
    const float* fc1_b = (const float*)d_in[4];
    const float* bn1_g = (const float*)d_in[5];
    const float* bn1_b = (const float*)d_in[6];
    const float* bn1_m = (const float*)d_in[7];
    const float* bn1_v = (const float*)d_in[8];
    const float* gc_w  = (const float*)d_in[9];
    const float* gc_b  = (const float*)d_in[10];
    const float* bng_g = (const float*)d_in[11];
    const float* bng_b = (const float*)d_in[12];
    const float* bng_m = (const float*)d_in[13];
    const float* bng_v = (const float*)d_in[14];
    const float* fc2_w = (const float*)d_in[15];
    const float* fc2_b = (const float*)d_in[16];
    const float* bn2_g = (const float*)d_in[17];
    const float* bn2_b = (const float*)d_in[18];
    const float* bn2_m = (const float*)d_in[19];
    const float* bn2_v = (const float*)d_in[20];

    const size_t NELEM = (size_t)32 * CCH * HWP;   // 19,267,584

    // workspace carve-out (256B aligned)
    char* ws = (char*)d_ws;
    size_t off = 0;
    auto carve = [&](size_t bytes) -> char* {
        char* p = ws + off;
        off = (off + bytes + 255) & ~(size_t)255;
        return p;
    };
    unsigned short* wb    = (unsigned short*)carve(147456 * 2);   // bf16 weights
    float*          s_all = (float*)carve(576 * 4);
    float*          t_all = (float*)carve(576 * 4);
    unsigned short* ybf   = (unsigned short*)carve(NELEM * 2);    // NHWC
    unsigned short* x1bf  = (unsigned short*)carve(NELEM * 2);    // NHWC
    unsigned short* x2bf  = (unsigned short*)carve(NELEM * 2);    // NHWC
    (void)ws_size;

    cvt_weights<<<(147456 + 255) / 256, 256, 0, stream>>>(fc1_w, gc_w, fc2_w, wb);
    bn_consts<<<1, 576, 0, stream>>>(fc1_b, bn1_g, bn1_b, bn1_m, bn1_v,
                                     gc_b,  bng_g, bng_b, bng_m, bng_v,
                                     fc2_b, bn2_g, bn2_b, bn2_m, bn2_v,
                                     s_all, t_all);
    dwconv7<<<(int)((NELEM + 255) / 256), 256, 0, stream>>>(x, cpe_w, cpe_b, ybf, (int)NELEM);

    dim3 grid(HWP / 64, CCH / 64, 32);   // (49, 3, 32)
    gemm1x1<0, CCH>    <<<grid, 128, 0, stream>>>(wb,                 ybf,
                                                  s_all,       t_all, x1bf, nullptr);
    gemm1x1<1, 2 * CCH><<<grid, 128, 0, stream>>>(wb + 36864,         x1bf,
                                                  s_all + 192, t_all + 192, x2bf, nullptr);
    gemm1x1<2, CCH>    <<<grid, 128, 0, stream>>>(wb + 36864 + 73728, x2bf,
                                                  s_all + 384, t_all + 384, nullptr,
                                                  (float*)d_out);
}
